// Macom_29583734734848
// MI455X (gfx1250) — matrix operand order repaired and checked by reading, not compiled, not run on hardware
//
#include <hip/hip_runtime.h>

// ---------------------------------------------------------------------------
// Problem constants
// ---------------------------------------------------------------------------
#define BB   512   // batches
#define NA   256   // agents (N)
#define MSGK 256   // message dim (K of encoder)
#define DD   512   // hidden dim
#define OUTD 256   // output dim

typedef __attribute__((ext_vector_type(16))) __bf16 v16bf;
typedef __attribute__((ext_vector_type(8)))  float  v8f;

union FragAB { v16bf v; uint4 q[2]; __bf16 h[16]; };
union Acc8   { v8f v; float f[8]; };

__device__ __forceinline__ v8f wmma_bf16(v16bf a, v16bf b, v8f c) {
  // D = A(16x32 bf16) * B(32x16 bf16) + C(16x16 f32)
  return __builtin_amdgcn_wmma_f32_16x16x32_bf16(
      /*neg_a=*/false, a, /*neg_b=*/false, b,
      /*c_mod=*/(short)0, c, /*reuse_a=*/false, /*reuse_b=*/false);
}

// A fragment 16x32: lane m = ln, K segments {k0+half*8 .. +8} and {k0+16+half*8 .. +8}
__device__ __forceinline__ v16bf load_fragA(const __bf16* src, int stride,
                                            int r0, int k0, int half, int ln) {
  FragAB f;
  const __bf16* p = src + (size_t)(r0 + ln) * stride + (k0 + half * 8);
  f.q[0] = *(const uint4*)(p);
  f.q[1] = *(const uint4*)(p + 16);
  return f.v;
}

// B fragment 32x16 from "n-major" source src[n][k]: lane n = ln, K segments {k0+half*8 .. +8} and {k0+16+half*8 .. +8}
__device__ __forceinline__ v16bf load_fragB(const __bf16* src, int stride,
                                            int n0, int k0, int half, int ln) {
  FragAB f;
  const __bf16* p = src + (size_t)(n0 + ln) * stride + (k0 + half * 8);
  f.q[0] = *(const uint4*)(p);
  f.q[1] = *(const uint4*)(p + 16);
  return f.v;
}

// A fragment from f32 source, converting to bf16 in-register
__device__ __forceinline__ v16bf load_fragA_f32(const float* src, int stride,
                                                int r0, int k0, int half, int ln) {
  FragAB f;
  const float* p = src + (size_t)(r0 + ln) * stride + (k0 + half * 8);
#pragma unroll
  for (int i = 0; i < 8; ++i) f.h[i] = (__bf16)p[i];
#pragma unroll
  for (int i = 0; i < 8; ++i) f.h[8 + i] = (__bf16)p[16 + i];
  return f.v;
}

// ---------------------------------------------------------------------------
// Kernel 0: transpose + downconvert weights to bf16 (B-fragment-friendly)
//   W_encT[d][m] = W_enc[m][d]   (DD x MSGK)
//   W_decT[o][d] = W_dec[d][o]   (OUTD x DD)
// ---------------------------------------------------------------------------
__global__ void prep_weights(const float* __restrict__ W_enc,
                             const float* __restrict__ W_dec,
                             __bf16* __restrict__ WeT,
                             __bf16* __restrict__ WdT) {
  int idx = blockIdx.x * blockDim.x + threadIdx.x;
  if (idx < MSGK * DD) {
    int m = idx / DD, d = idx % DD;
    WeT[(size_t)d * MSGK + m] = (__bf16)W_enc[idx];
  }
  if (idx < DD * OUTD) {
    int d = idx / OUTD, o = idx % OUTD;
    WdT[(size_t)o * DD + d] = (__bf16)W_dec[idx];
  }
}

// ---------------------------------------------------------------------------
// Kernel 1: enc = x @ W_enc + b_enc  (M=131072, K=256, N=512), bf16 out.
// Writes enc [B*N][D] and encT [B][D][N] (both bf16).
// Block = 8 waves; block tile = 128 rows x 128 cols; wave = 16 rows x 128 cols.
// ---------------------------------------------------------------------------
__global__ __launch_bounds__(256, 2)
void enc_kernel(const float* __restrict__ x,
                const __bf16* __restrict__ WeT,
                const float* __restrict__ b_enc,
                __bf16* __restrict__ enc_g,
                __bf16* __restrict__ encT_g) {
  const int w = threadIdx.x >> 5;
  const int lane = threadIdx.x & 31;
  const int half = lane >> 4, ln = lane & 15;

  const int r0 = blockIdx.x * 128 + w * 16;   // global row (b*N + n)
  const int d_base = blockIdx.y * 128;

  Acc8 acc[8];
#pragma unroll
  for (int ct = 0; ct < 8; ++ct)
#pragma unroll
    for (int i = 0; i < 8; ++i) acc[ct].f[i] = 0.0f;

  for (int kk = 0; kk < MSGK; kk += 32) {
    v16bf av = load_fragA_f32(x, MSGK, r0, kk, half, ln);
#pragma unroll
    for (int ct = 0; ct < 8; ++ct) {
      v16bf bv = load_fragB(WeT, MSGK, d_base + ct * 16, kk, half, ln);
      acc[ct].v = wmma_bf16(av, bv, acc[ct].v);
    }
  }

  const int bb = r0 >> 8;               // batch
  const int n0 = (r0 & 255) + half * 8; // agent index base for this lane's rows
#pragma unroll
  for (int ct = 0; ct < 8; ++ct) {
    const int d = d_base + ct * 16 + ln;
    const float bias = b_enc[d];
    union { uint4 q; __bf16 h[8]; } pk;
#pragma unroll
    for (int v = 0; v < 8; ++v) {
      float val = acc[ct].f[v] + bias;
      pk.h[v] = (__bf16)val;
      // enc: row-major [row][d] (scattered b16 stores, column per lane)
      enc_g[(size_t)(r0 + half * 8 + v) * DD + d] = pk.h[v];
    }
    // encT: [b][d][n] -> this lane's 8 values are contiguous: one b128 store
    *(uint4*)(encT_g + ((size_t)bb * DD + d) * NA + n0) = pk.q;
  }
}

// ---------------------------------------------------------------------------
// Kernel 2: fused attention per batch.
//   S = enc enc^T ; wn = softmax_axis1(S) ; P = wn enc ; out_b = max_rows(P W_dec) + b_dec
// One workgroup (16 waves, 512 threads) per batch.
// Dynamic LDS: [0,256K) enc/encT (union), [256K,288K) scratch (aliased with
// reduction partials), [288K,+2K) colmax/colinv.
// ---------------------------------------------------------------------------
#define SMEM_MAIN   (NA * DD * 2)         // 262144
#define SMEM_SCR    (16 * 16 * 64 * 2)    // 32768 (2KB per wave)
#define SMEM_SMALL  (2 * NA * 4)          // 2048
#define SMEM_TOTAL  (SMEM_MAIN + SMEM_SCR + SMEM_SMALL)

__global__ __launch_bounds__(512, 1)
void attn_kernel(const __bf16* __restrict__ enc_g,
                 const __bf16* __restrict__ encT_g,
                 const __bf16* __restrict__ WdT,
                 const float* __restrict__ b_dec,
                 float* __restrict__ out) {
  extern __shared__ __align__(16) char smem[];
  __bf16* sMain    = (__bf16*)smem;                   // enc [NA][DD] then encT [DD][NA]
  __bf16* sScr     = (__bf16*)(smem + SMEM_MAIN);     // per-wave transpose scratch
  float*  sPart    = (float*)(smem + SMEM_MAIN);      // aliased [16][NA] reduction partials
  float*  sColMax  = (float*)(smem + SMEM_MAIN + SMEM_SCR);
  float*  sColInv  = sColMax + NA;

  const int b    = blockIdx.x;
  const int tid  = threadIdx.x;
  const int w    = tid >> 5;
  const int lane = tid & 31;
  const int half = lane >> 4, ln = lane & 15;
  const int i0   = w * 16;                 // this wave's row strip
  __bf16* myScr  = sScr + w * 1024;        // 16x64 bf16 per wave

  // ---- stage enc[b] (256KB) into LDS --------------------------------------
  {
    const uint4* src = (const uint4*)(enc_g + (size_t)b * NA * DD);
    uint4* dst = (uint4*)sMain;
    for (int i = tid; i < (NA * DD * 2) / 16; i += 512) dst[i] = src[i];
  }
  __syncthreads();

  // ---- Phase S: S = enc enc^T (this wave: rows i0..i0+15, all 256 cols) ---
  Acc8 sacc[16];
#pragma unroll
  for (int jt = 0; jt < 16; ++jt)
#pragma unroll
    for (int i = 0; i < 8; ++i) sacc[jt].f[i] = 0.0f;

  for (int kk = 0; kk < DD; kk += 32) {
    v16bf av = load_fragA(sMain, DD, i0, kk, half, ln);
#pragma unroll
    for (int jt = 0; jt < 16; ++jt) {
      v16bf bv = load_fragB(sMain, DD, jt * 16, kk, half, ln);
      sacc[jt].v = wmma_bf16(av, bv, sacc[jt].v);
    }
  }

  // ---- column softmax (normalize over rows i, per column j) ---------------
  // column max: local over 8 rows, fold halves with shfl_xor(16), cross-wave via LDS
#pragma unroll
  for (int jt = 0; jt < 16; ++jt) {
    float m = sacc[jt].f[0];
#pragma unroll
    for (int v = 1; v < 8; ++v) m = fmaxf(m, sacc[jt].f[v]);
    m = fmaxf(m, __shfl_xor(m, 16));
    if (half == 0) sPart[w * NA + jt * 16 + ln] = m;
  }
  __syncthreads();
  if (tid < NA) {
    float m = -3.0e38f;
#pragma unroll
    for (int ww = 0; ww < 16; ++ww) m = fmaxf(m, sPart[ww * NA + tid]);
    sColMax[tid] = m;
  }
  __syncthreads();
  // exp + column sum
#pragma unroll
  for (int jt = 0; jt < 16; ++jt) {
    const float cm = sColMax[jt * 16 + ln];
    float s = 0.0f;
#pragma unroll
    for (int v = 0; v < 8; ++v) {
      float e = __expf(sacc[jt].f[v] - cm);
      sacc[jt].f[v] = e;
      s += e;
    }
    s += __shfl_xor(s, 16);
    if (half == 0) sPart[w * NA + jt * 16 + ln] = s;
  }
  __syncthreads();
  if (tid < NA) {
    float s = 0.0f;
#pragma unroll
    for (int ww = 0; ww < 16; ++ww) s += sPart[ww * NA + tid];
    sColInv[tid] = 1.0f / s;
  }
  __syncthreads();
#pragma unroll
  for (int jt = 0; jt < 16; ++jt) {
    const float inv = sColInv[jt * 16 + ln];
#pragma unroll
    for (int v = 0; v < 8; ++v) sacc[jt].f[v] *= inv;
  }

  // ---- stage encT[b] (overwrites enc region; enc fully consumed above) ----
  __syncthreads();
  {
    const uint4* src = (const uint4*)(encT_g + (size_t)b * DD * NA);
    uint4* dst = (uint4*)sMain;
    for (int i = tid; i < (NA * DD * 2) / 16; i += 512) dst[i] = src[i];
  }
  __syncthreads();

  // ---- wn: C-layout f32 -> A-layout bf16 via per-wave LDS scratch ---------
  // (same-wave DS store->load is in-order; no cross-wave sharing -> no barrier)
  v16bf wnA[8];
  for (int jc = 0; jc < NA; jc += 32) {
#pragma unroll
    for (int t = 0; t < 2; ++t) {
      const int jt = (jc >> 4) + t;
#pragma unroll
      for (int v = 0; v < 8; ++v)
        myScr[(half * 8 + v) * 32 + t * 16 + ln] = (__bf16)sacc[jt].f[v];
    }
    wnA[jc >> 5] = load_fragA(myScr, 32, 0, 0, half, ln);
  }

  // ---- Phase P + decode fused over 64-wide d-chunks -----------------------
  Acc8 outacc[16];
#pragma unroll
  for (int ot = 0; ot < 16; ++ot)
#pragma unroll
    for (int i = 0; i < 8; ++i) outacc[ot].f[i] = 0.0f;

  for (int dc = 0; dc < DD; dc += 64) {
    Acc8 pacc[4];
#pragma unroll
    for (int dt = 0; dt < 4; ++dt)
#pragma unroll
      for (int i = 0; i < 8; ++i) pacc[dt].f[i] = 0.0f;

    for (int jc = 0; jc < NA; jc += 32) {
#pragma unroll
      for (int dt = 0; dt < 4; ++dt) {
        // B[k=j][n=d] from encT[d][j] (contiguous along j per lane)
        v16bf bv = load_fragB(sMain, NA, dc + dt * 16, jc, half, ln);
        pacc[dt].v = wmma_bf16(wnA[jc >> 5], bv, pacc[dt].v);
      }
    }

    // P chunk: C-layout f32 -> A-layout bf16 via per-wave scratch (16x64)
#pragma unroll
    for (int dt = 0; dt < 4; ++dt)
#pragma unroll
      for (int v = 0; v < 8; ++v)
        myScr[(half * 8 + v) * 64 + dt * 16 + ln] = (__bf16)pacc[dt].f[v];
    v16bf pA0 = load_fragA(myScr, 64, 0, 0, half, ln);
    v16bf pA1 = load_fragA(myScr, 64, 0, 32, half, ln);

    // decode: out += P_chunk @ W_dec_chunk   (B-frags from W_decT[o][d], L2-hot)
#pragma unroll
    for (int ot = 0; ot < 16; ++ot) {
      v16bf b0 = load_fragB(WdT, DD, ot * 16, dc, half, ln);
      v16bf b1 = load_fragB(WdT, DD, ot * 16, dc + 32, half, ln);
      outacc[ot].v = wmma_bf16(pA0, b0, outacc[ot].v);
      outacc[ot].v = wmma_bf16(pA1, b1, outacc[ot].v);
    }
  }

  // ---- column max over all 256 rows, + b_dec, store out[b][:] -------------
  __syncthreads();  // scratch region reused as sPart below
#pragma unroll
  for (int ot = 0; ot < 16; ++ot) {
    float m = outacc[ot].f[0];
#pragma unroll
    for (int v = 1; v < 8; ++v) m = fmaxf(m, outacc[ot].f[v]);
    m = fmaxf(m, __shfl_xor(m, 16));
    if (half == 0) sPart[w * NA + ot * 16 + ln] = m;
  }
  __syncthreads();
  if (tid < OUTD) {
    float m = -3.0e38f;
#pragma unroll
    for (int ww = 0; ww < 16; ++ww) m = fmaxf(m, sPart[ww * NA + tid]);
    out[(size_t)b * OUTD + tid] = m + b_dec[tid];
  }
}

// ---------------------------------------------------------------------------
// Launch
// ---------------------------------------------------------------------------
extern "C" void kernel_launch(void* const* d_in, const int* in_sizes, int n_in,
                              void* d_out, int out_size, void* d_ws, size_t ws_size,
                              hipStream_t stream) {
  const float* x     = (const float*)d_in[0];   // [B,N,MSG]
  const float* W_enc = (const float*)d_in[1];   // [MSG,D]
  const float* b_enc = (const float*)d_in[2];   // [D]
  const float* W_dec = (const float*)d_in[3];   // [D,OUT]
  const float* b_dec = (const float*)d_in[4];   // [OUT]
  float* out = (float*)d_out;                   // [B,OUT]

  char* ws = (char*)d_ws;
  const size_t encBytes = (size_t)BB * NA * DD * 2;          // 134217728
  __bf16* enc_g  = (__bf16*)ws;
  __bf16* encT_g = (__bf16*)(ws + encBytes);
  __bf16* WeT    = (__bf16*)(ws + 2 * encBytes);
  __bf16* WdT    = (__bf16*)(ws + 2 * encBytes + (size_t)DD * MSGK * 2);

  prep_weights<<<(MSGK * DD + 255) / 256, 256, 0, stream>>>(W_enc, W_dec, WeT, WdT);

  dim3 gE((BB * NA) / 128, DD / 128);
  enc_kernel<<<gE, 256, 0, stream>>>(x, WeT, b_enc, enc_g, encT_g);

  attn_kernel<<<BB, 512, SMEM_TOTAL, stream>>>(enc_g, encT_g, WdT, b_dec, out);
}
